// RecurrentLinearAttention_8117488189602
// MI455X (gfx1250) — matrix operand order repaired and checked by reading, not compiled, not run on hardware
//
#include <hip/hip_runtime.h>
#include <hip/hip_fp16.h>

typedef __attribute__((ext_vector_type(16))) _Float16 v16h;
typedef __attribute__((ext_vector_type(8)))  float    v8f;

#define NTOK 2048
#define DIM  512
#define NH   8
#define Dh   64
#define EPS  1e-6f

// ---------------------------------------------------------------------------
// Kernel 1: QKV projection GEMM with fused bias + (elu+1) activation.
//   dst[n][j] = act( sum_k X[n][k] * W[j][k] + b[j] )
// Both X [N,K] and W [J,K] are row-major with K contiguous -> A @ B^T shape,
// ideal for WMMA: A rows and B "columns" are both contiguous K-runs.
// Block = 256 threads = 8 waves; block tile = 128 rows x 64 cols.
// Each wave: 64 rows x 16 cols = 4x v_wmma_f32_16x16x32_f16 accumulators,
// B fragment reused across the 4 row tiles. grid.z in {0,1,2} = Q,K,V.
// ---------------------------------------------------------------------------
__global__ __launch_bounds__(256) void qkv_gemm_kernel(
    const float* __restrict__ q_in, const float* __restrict__ k_in,
    const float* __restrict__ v_in,
    const float* __restrict__ Wq, const float* __restrict__ bq,
    const float* __restrict__ Wk, const float* __restrict__ bk,
    const float* __restrict__ Wv, const float* __restrict__ bv,
    float* __restrict__ ws)
{
    const int z = blockIdx.z;
    const float* X = (z == 0) ? q_in : (z == 1) ? k_in : v_in;
    const float* W = (z == 0) ? Wq   : (z == 1) ? Wk   : Wv;
    const float* B = (z == 0) ? bq   : (z == 1) ? bk   : bv;
    float* dst = ws + (size_t)z * ((size_t)NTOK * DIM);

    const int wave   = threadIdx.x >> 5;
    const int lane   = threadIdx.x & 31;
    const int lanelo = lane & 15;
    const int hi     = (lane >= 16) ? 1 : 0;

    const int rowbase = blockIdx.x * 128 + (wave >> 2) * 64;
    const int col     = blockIdx.y * 64  + (wave & 3) * 16 + lanelo;

    v8f acc[4];
    acc[0] = (v8f){}; acc[1] = (v8f){}; acc[2] = (v8f){}; acc[3] = (v8f){};

    // B-matrix (32x16, K-major per lane): lane = N index, lanes 0-15 hold
    // K = kb+0..15, lanes 16-31 hold K = kb+16..31 (16 consecutive halves).
    const float* wrow = W + (size_t)col * DIM + (hi ? 16 : 0);
    // A-matrix (16x32): lane row = rowbase + lanelo; lanes 0-15 hold
    // K = kb+0..15; lanes 16-31 hold K = kb+16..31.
    const float* xrow = X + (size_t)(rowbase + lanelo) * DIM + (hi ? 16 : 0);

    for (int kb = 0; kb < DIM; kb += 32) {
        v16h bf;
        {
            const float4* p = (const float4*)(wrow + kb);
            float4 f0 = p[0], f1 = p[1], f2 = p[2], f3 = p[3];
            bf[0]  = (_Float16)f0.x; bf[1]  = (_Float16)f0.y;
            bf[2]  = (_Float16)f0.z; bf[3]  = (_Float16)f0.w;
            bf[4]  = (_Float16)f1.x; bf[5]  = (_Float16)f1.y;
            bf[6]  = (_Float16)f1.z; bf[7]  = (_Float16)f1.w;
            bf[8]  = (_Float16)f2.x; bf[9]  = (_Float16)f2.y;
            bf[10] = (_Float16)f2.z; bf[11] = (_Float16)f2.w;
            bf[12] = (_Float16)f3.x; bf[13] = (_Float16)f3.y;
            bf[14] = (_Float16)f3.z; bf[15] = (_Float16)f3.w;
        }
        #pragma unroll
        for (int mt = 0; mt < 4; ++mt) {
            const float* xp = xrow + (size_t)mt * 16 * DIM + kb;
            const float4* plo = (const float4*)(xp);        // K+0..7  (or +16..23)
            const float4* phi = (const float4*)(xp + 8);    // K+8..15 (or +24..31)
            float4 a0 = plo[0], a1 = plo[1], a2 = phi[0], a3 = phi[1];
            v16h af;
            af[0]  = (_Float16)a0.x; af[1]  = (_Float16)a0.y;
            af[2]  = (_Float16)a0.z; af[3]  = (_Float16)a0.w;
            af[4]  = (_Float16)a1.x; af[5]  = (_Float16)a1.y;
            af[6]  = (_Float16)a1.z; af[7]  = (_Float16)a1.w;
            af[8]  = (_Float16)a2.x; af[9]  = (_Float16)a2.y;
            af[10] = (_Float16)a2.z; af[11] = (_Float16)a2.w;
            af[12] = (_Float16)a3.x; af[13] = (_Float16)a3.y;
            af[14] = (_Float16)a3.z; af[15] = (_Float16)a3.w;
            acc[mt] = __builtin_amdgcn_wmma_f32_16x16x32_f16(
                false, af, false, bf, (short)0, acc[mt], false, false);
        }
    }

    // Epilogue: C/D layout -> lane L, elem r: row = r + (L<16?0:8), col = L&15.
    const float bj = B[col];
    #pragma unroll
    for (int mt = 0; mt < 4; ++mt) {
        #pragma unroll
        for (int r = 0; r < 8; ++r) {
            int row = rowbase + mt * 16 + r + (hi ? 8 : 0);
            float v = acc[mt][r] + bj;
            if (z != 2) v = (v > 0.0f) ? (v + 1.0f) : __expf(v);  // elu(v)+1
            dst[(size_t)row * DIM + col] = v;
        }
    }
}

// ---------------------------------------------------------------------------
// Kernel 2: per-(n,h) state update (memory-bound: 16 KB Si in + 16 KB out).
//   Zi_new = Zi + K ; Si_new = Si + K (x) V
//   out = (Q . Si_new) / (Q . Zi_new + eps)
// One block per (n,h). Thread t: rows rg*4..rg*4+3 (rg=t>>4),
// cols cg*4..cg*4+3 (cg=t&15); float4 streaming, LDS reduction over rows.
// ---------------------------------------------------------------------------
__global__ __launch_bounds__(256) void state_update_kernel(
    const float* __restrict__ ws, const float* __restrict__ Si,
    const float* __restrict__ Zi,
    float* __restrict__ out, float* __restrict__ Si_new,
    float* __restrict__ Zi_new)
{
    __shared__ float sQ[Dh], sK[Dh], sV[Dh], sdot[Dh];
    __shared__ float sred[16 * Dh];
    __shared__ float sZinv;

    const int b = blockIdx.x;          // n * NH + h
    const int t = threadIdx.x;
    const float* Qp = ws;
    const float* Kp = ws + (size_t)NTOK * DIM;
    const float* Vp = ws + 2 * (size_t)NTOK * DIM;
    const size_t vecoff = (size_t)b * Dh;        // == n*DIM + h*Dh (flat [N,H,D])
    const size_t soff   = (size_t)b * (Dh * Dh); // flat [N,H,D,M]

    if (t < Dh) {
        float q  = Qp[vecoff + t];
        float k  = Kp[vecoff + t];
        float zn = Zi[vecoff + t] + k;
        Zi_new[vecoff + t] = zn;
        sQ[t] = q; sK[t] = k; sdot[t] = q * zn;
    } else if (t < 2 * Dh) {
        int d = t - Dh;
        sV[d] = Vp[vecoff + d];
    }
    __syncthreads();

    if (t == 0) {
        float s = 0.0f;
        #pragma unroll
        for (int i = 0; i < Dh; ++i) s += sdot[i];
        sZinv = 1.0f / (s + EPS);
    }
    __syncthreads();

    const int rg = t >> 4;   // 0..15 : row group (4 rows)
    const int cg = t & 15;   // 0..15 : col group (4 cols)
    const float* sip = Si + soff;
    float* snp = Si_new + soff;
    const float4 v4 = *(const float4*)(sV + cg * 4);

    float4 acc = make_float4(0.f, 0.f, 0.f, 0.f);
    #pragma unroll
    for (int r = 0; r < 4; ++r) {
        const int d  = rg * 4 + r;
        const float kd = sK[d];
        const float qd = sQ[d];
        float4 s = *(const float4*)(sip + d * Dh + cg * 4);
        float4 sn;
        sn.x = fmaf(kd, v4.x, s.x);
        sn.y = fmaf(kd, v4.y, s.y);
        sn.z = fmaf(kd, v4.z, s.z);
        sn.w = fmaf(kd, v4.w, s.w);
        *(float4*)(snp + d * Dh + cg * 4) = sn;
        acc.x = fmaf(qd, sn.x, acc.x);
        acc.y = fmaf(qd, sn.y, acc.y);
        acc.z = fmaf(qd, sn.z, acc.z);
        acc.w = fmaf(qd, sn.w, acc.w);
    }
    {
        float* pr = sred + rg * Dh + cg * 4;
        pr[0] = acc.x; pr[1] = acc.y; pr[2] = acc.z; pr[3] = acc.w;
    }
    __syncthreads();

    if (t < Dh) {
        float tot = 0.0f;
        #pragma unroll
        for (int rg2 = 0; rg2 < 16; ++rg2) tot += sred[rg2 * Dh + t];
        out[vecoff + t] = tot * sZinv;
    }
}

extern "C" void kernel_launch(void* const* d_in, const int* in_sizes, int n_in,
                              void* d_out, int out_size, void* d_ws, size_t ws_size,
                              hipStream_t stream) {
    const float* query = (const float*)d_in[0];
    const float* key   = (const float*)d_in[1];
    const float* value = (const float*)d_in[2];
    const float* Si    = (const float*)d_in[3];
    const float* Zi    = (const float*)d_in[4];
    const float* Wq    = (const float*)d_in[5];
    const float* bq    = (const float*)d_in[6];
    const float* Wk    = (const float*)d_in[7];
    const float* bk    = (const float*)d_in[8];
    const float* Wv    = (const float*)d_in[9];
    const float* bv    = (const float*)d_in[10];

    float* out    = (float*)d_out;
    float* Si_new = out + (size_t)NTOK * DIM;
    float* Zi_new = Si_new + (size_t)NTOK * NH * Dh * Dh;
    float* ws     = (float*)d_ws;   // needs 3 * N * DIM * 4 bytes = 12 MB

    dim3 g1(NTOK / 128, DIM / 64, 3);
    qkv_gemm_kernel<<<g1, 256, 0, stream>>>(query, key, value,
                                            Wq, bq, Wk, bk, Wv, bv, ws);

    state_update_kernel<<<NTOK * NH, 256, 0, stream>>>(ws, Si, Zi,
                                                       out, Si_new, Zi_new);
}
